// ToyL3Model_83631603187733
// MI455X (gfx1250) — compile-verified, matrix operand order checked by reading, not run on hardware
//
#include <hip/hip_runtime.h>
#include <hip/hip_bf16.h>

// ---------------------------------------------------------------------------
// Problem constants (from the reference)
// ---------------------------------------------------------------------------
#define MTOK 4096   // B*T = 4*1024
#define DM   2048   // d_model
#define DFF  8192   // d_ff
#define NKV  256    // n_kv
#define SSEL 16     // s (selected kv per token)

typedef __attribute__((ext_vector_type(16))) __bf16 v16bf;
typedef __attribute__((ext_vector_type(8)))  float  v8f;
typedef __attribute__((ext_vector_type(4)))  int    v4i;

// ---------------------------------------------------------------------------
// CDNA5 async global->LDS path (guarded; falls back to VGPR staging)
// ---------------------------------------------------------------------------
#if defined(__gfx1250__) && \
    __has_builtin(__builtin_amdgcn_global_load_async_to_lds_b128) && \
    __has_builtin(__builtin_amdgcn_s_wait_asynccnt)
#define USE_ASYNC_LDS 1
#else
#define USE_ASYNC_LDS 0
#endif

#if USE_ASYNC_LDS
typedef __attribute__((address_space(1))) v4i* as1_v4i_p;   // global v4i32*
typedef __attribute__((address_space(3))) v4i* as3_v4i_p;   // LDS v4i32*

__device__ __forceinline__ void async_copy_b128(const unsigned short* g, unsigned short* l) {
    // generic -> AS pointers via integer round-trip (low 32 bits of a flat LDS
    // pointer are the group-segment offset per the CDNA5 aperture mapping).
    __builtin_amdgcn_global_load_async_to_lds_b128(
        (as1_v4i_p)(unsigned long long)(const void*)g,
        (as3_v4i_p)(unsigned int)(unsigned long long)(const void*)l,
        0, 0);
}
__device__ __forceinline__ void wait_async0() { __builtin_amdgcn_s_wait_asynccnt(0); }
#endif

// ---------------------------------------------------------------------------
// bf16 helpers (RNE)
// ---------------------------------------------------------------------------
__device__ __forceinline__ unsigned short f2bf(float f) {
    unsigned int u = __float_as_uint(f);
    unsigned int r = u + 0x7FFFu + ((u >> 16) & 1u);
    return (unsigned short)(r >> 16);
}
__device__ __forceinline__ float bf2f(unsigned short h) {
    return __uint_as_float(((unsigned int)h) << 16);
}

// ---------------------------------------------------------------------------
// Scale slots init
// ---------------------------------------------------------------------------
__global__ void init_slots(unsigned int* slots) {
    if (threadIdx.x < 4) slots[threadIdx.x] = 0u;
}

// ---------------------------------------------------------------------------
// max|w| reduction -> atomicMax on uint bits (valid ordering: values >= 0)
// ---------------------------------------------------------------------------
__global__ __launch_bounds__(256) void maxabs_kernel(const float* __restrict__ w,
                                                     size_t n, unsigned int* slot) {
    __shared__ float red[256];
    float m = 0.0f;
    for (size_t i = (size_t)blockIdx.x * blockDim.x + threadIdx.x; i < n;
         i += (size_t)gridDim.x * blockDim.x)
        m = fmaxf(m, fabsf(w[i]));
    red[threadIdx.x] = m;
    __syncthreads();
    for (int s = 128; s > 0; s >>= 1) {
        if ((int)threadIdx.x < s) red[threadIdx.x] = fmaxf(red[threadIdx.x], red[threadIdx.x + s]);
        __syncthreads();
    }
    if (threadIdx.x == 0) atomicMax(slot, __float_as_uint(red[0]));
}

// ---------------------------------------------------------------------------
// Fake-W8 quantize -> bf16 (same layout)
// ---------------------------------------------------------------------------
__global__ __launch_bounds__(256) void quant_bf16(const float* __restrict__ w,
                                                  unsigned short* __restrict__ out,
                                                  size_t n, const unsigned int* slot) {
    const float mx    = __uint_as_float(*slot);
    const float scale = 127.0f / fmaxf(mx, 1e-8f);
    const float inv   = 1.0f / scale;
    for (size_t i = (size_t)blockIdx.x * blockDim.x + threadIdx.x; i < n;
         i += (size_t)gridDim.x * blockDim.x) {
        float q = fminf(fmaxf(rintf(w[i] * scale), -128.0f), 127.0f) * inv;
        out[i] = f2bf(q);
    }
}

// W_V [DM, NKV] -> quantized transposed bf16 [NKV, DM]
__global__ __launch_bounds__(256) void quant_bf16_transpose(const float* __restrict__ w,
                                                            unsigned short* __restrict__ out,
                                                            const unsigned int* slot) {
    const float mx    = __uint_as_float(*slot);
    const float scale = 127.0f / fmaxf(mx, 1e-8f);
    const float inv   = 1.0f / scale;
    const size_t n = (size_t)DM * NKV;
    for (size_t i = (size_t)blockIdx.x * blockDim.x + threadIdx.x; i < n;
         i += (size_t)gridDim.x * blockDim.x) {
        size_t d = i / NKV, k = i % NKV;
        float q = fminf(fmaxf(rintf(w[i] * scale), -128.0f), 127.0f) * inv;
        out[k * DM + d] = f2bf(q);
    }
}

// f32 -> bf16 elementwise
__global__ __launch_bounds__(256) void cvt_bf16(const float* __restrict__ in,
                                                unsigned short* __restrict__ out, size_t n) {
    for (size_t i = (size_t)blockIdx.x * blockDim.x + threadIdx.x; i < n;
         i += (size_t)gridDim.x * blockDim.x)
        out[i] = f2bf(in[i]);
}

// ---------------------------------------------------------------------------
// Per-token softmax over 16 gathered scores; scatter-add probs into dense
// P[tok, NKV] (bf16). Duplicate indices accumulate exactly like the reference.
// ---------------------------------------------------------------------------
__global__ __launch_bounds__(256) void attn_softmax_scatter(const float* __restrict__ S,
                                                            const int* __restrict__ idx,
                                                            unsigned short* __restrict__ P) {
    int t = blockIdx.x * blockDim.x + threadIdx.x;
    if (t >= MTOK) return;
    unsigned short* prow = P + (size_t)t * NKV;
    uint4* p4 = reinterpret_cast<uint4*>(prow);
    uint4 z; z.x = z.y = z.z = z.w = 0u;
#pragma unroll
    for (int i = 0; i < (NKV * 2) / 16; ++i) p4[i] = z;   // zero 512 bytes

    const float* srow = S + (size_t)t * NKV;
    const int*   irow = idx + t * SSEL;
    int   id[SSEL];
    float sc[SSEL];
    float mx = -3.4e38f;
#pragma unroll
    for (int s = 0; s < SSEL; ++s) {
        id[s] = irow[s];
        sc[s] = srow[id[s]];
        mx = fmaxf(mx, sc[s]);
    }
    float sum = 0.0f;
#pragma unroll
    for (int s = 0; s < SSEL; ++s) { sc[s] = __expf(sc[s] - mx); sum += sc[s]; }
    float inv = 1.0f / sum;
#pragma unroll
    for (int s = 0; s < SSEL; ++s) {
        int k = id[s];
        prow[k] = f2bf(bf2f(prow[k]) + sc[s] * inv);      // sequential per-thread RMW
    }
}

// ---------------------------------------------------------------------------
// WMMA bf16 GEMM: C[M,N] = A[M,K] * B[K,N]; A,B row-major bf16, C f32 or bf16.
// 256 threads = 8 waves; 128x128 C tile; K stepped by 32 through DOUBLE-
// BUFFERED LDS. Per wave: 64x32 subtile = 4x2 v_wmma_f32_16x16x32_bf16 accums.
// A tile staged via GLOBAL_LOAD_ASYNC_TO_LDS_B128 (ASYNCcnt) when available.
// ---------------------------------------------------------------------------
__device__ __forceinline__ v16bf load_frag(const unsigned short* row, int hi) {
    struct alignas(32) Pack { uint4 lo, hi; } p;
    p.lo = *reinterpret_cast<const uint4*>(row + hi * 8);        // K = hi*8 .. +7
    p.hi = *reinterpret_cast<const uint4*>(row + 16 + hi * 8);   // K = 16+hi*8 .. +7
    return __builtin_bit_cast(v16bf, p);
}

template <bool OUT_BF16>
__global__ __launch_bounds__(256) void gemm_bf16_wmma(const unsigned short* __restrict__ A,
                                                      const unsigned short* __restrict__ Bm,
                                                      void* __restrict__ Cout,
                                                      int M, int N, int K) {
    constexpr int LDA = 40, LDB = 40;                        // +8 bf16 pad, 16B-aligned rows
    __shared__ alignas(16) unsigned short Asm[2][128 * LDA]; // 2 x 10 KB
    __shared__ alignas(16) unsigned short Bsm[2][128 * LDB]; // 2 x 10 KB

    const int tid  = threadIdx.x;
    const int lane = tid & 31;
    const int wave = tid >> 5;
    const int ln   = lane & 15;
    const int hi   = lane >> 4;
    const int m0   = blockIdx.y * 128;
    const int n0   = blockIdx.x * 128;
    const int wM   = (wave >> 2) * 64;    // 0,64
    const int wN   = (wave & 3) * 32;     // 0,32,64,96

    v8f c[4][2];
#pragma unroll
    for (int i = 0; i < 4; ++i)
#pragma unroll
        for (int j = 0; j < 2; ++j) c[i][j] = v8f{0.f, 0.f, 0.f, 0.f, 0.f, 0.f, 0.f, 0.f};

    const int aRow = tid >> 1, aOff = (tid & 1) * 16;        // A tile: 128 rows x 32 k
    const int bK   = tid >> 3, bOff = (tid & 7) * 16;        // B tile: 32 k-rows x 128 n

    const unsigned short* gA = A  + (size_t)(m0 + aRow) * K + aOff;
    const unsigned short* gB = Bm + (size_t)bK * N + n0 + bOff;

    // ---- prologue: stage K-tile 0 into buffer 0 ----
#if USE_ASYNC_LDS
    {
        unsigned short* lA = &Asm[0][0] + aRow * LDA + aOff;
        async_copy_b128(gA,     lA);
        async_copy_b128(gA + 8, lA + 8);
    }
#else
    {
        const uint4* pa = reinterpret_cast<const uint4*>(gA);
        unsigned short* lA = &Asm[0][0] + aRow * LDA + aOff;
        *reinterpret_cast<uint4*>(lA)     = pa[0];
        *reinterpret_cast<uint4*>(lA + 8) = pa[1];
    }
#endif
    {
        const uint4* pb = reinterpret_cast<const uint4*>(gB);
        unsigned short tmp[16];
        *reinterpret_cast<uint4*>(tmp)     = pb[0];
        *reinterpret_cast<uint4*>(tmp + 8) = pb[1];
#pragma unroll
        for (int e = 0; e < 16; ++e) Bsm[0][(bOff + e) * LDB + bK] = tmp[e];
    }

    const int nT = K >> 5;
    for (int t = 0; t < nT; ++t) {
        const int cur = t & 1, nxt = cur ^ 1;
#if USE_ASYNC_LDS
        wait_async0();                 // my async writes for tile t have landed
#endif
        __syncthreads();               // everyone's tile-t data visible; buf[nxt] free

        const bool more = (t + 1) < nT;   // uniform across the block
        uint4 b0, b1;
#if !USE_ASYNC_LDS
        uint4 a0, a1;
#endif
        if (more) {
            const unsigned short* gAn = gA + (t + 1) * 32;
            const unsigned short* gBn = gB + (size_t)(t + 1) * 32 * N;
#if USE_ASYNC_LDS
            unsigned short* lAn = &Asm[nxt][0] + aRow * LDA + aOff;
            async_copy_b128(gAn,     lAn);     // overlaps with WMMA below
            async_copy_b128(gAn + 8, lAn + 8);
#else
            const uint4* pa = reinterpret_cast<const uint4*>(gAn);
            a0 = pa[0]; a1 = pa[1];
#endif
            const uint4* pb = reinterpret_cast<const uint4*>(gBn);
            b0 = pb[0]; b1 = pb[1];
            if (t + 2 < nT) __builtin_prefetch(gBn + (size_t)32 * N, 0, 0);
        }

        // ---- matrix core work: 8x v_wmma_f32_16x16x32_bf16 on buffer `cur` ----
        v16bf af[4], bfb[2];
#pragma unroll
        for (int i = 0; i < 4; ++i) af[i]  = load_frag(&Asm[cur][0] + (wM + i * 16 + ln) * LDA, hi);
#pragma unroll
        for (int j = 0; j < 2; ++j) bfb[j] = load_frag(&Bsm[cur][0] + (wN + j * 16 + ln) * LDB, hi);
#pragma unroll
        for (int i = 0; i < 4; ++i)
#pragma unroll
            for (int j = 0; j < 2; ++j)
                c[i][j] = __builtin_amdgcn_wmma_f32_16x16x32_bf16(
                    false, af[i], false, bfb[j], (short)0, c[i][j], false, false);

        if (more) {
#if !USE_ASYNC_LDS
            unsigned short* lAn = &Asm[nxt][0] + aRow * LDA + aOff;
            *reinterpret_cast<uint4*>(lAn)     = a0;
            *reinterpret_cast<uint4*>(lAn + 8) = a1;
#endif
            unsigned short tmp[16];
            *reinterpret_cast<uint4*>(tmp)     = b0;
            *reinterpret_cast<uint4*>(tmp + 8) = b1;
#pragma unroll
            for (int e = 0; e < 16; ++e) Bsm[nxt][(bOff + e) * LDB + bK] = tmp[e];
        }
    }

    // ---- store C: VGPR r -> row r + 8*hi, col = ln (per ISA C/D layout) ----
#pragma unroll
    for (int i = 0; i < 4; ++i)
#pragma unroll
        for (int j = 0; j < 2; ++j)
#pragma unroll
            for (int r = 0; r < 8; ++r) {
                int row = m0 + wM + i * 16 + hi * 8 + r;
                int col = n0 + wN + j * 16 + ln;
                float v = c[i][j][r];
                if (OUT_BF16)
                    reinterpret_cast<unsigned short*>(Cout)[(size_t)row * N + col] = f2bf(v);
                else
                    reinterpret_cast<float*>(Cout)[(size_t)row * N + col] = v;
            }
}

// ---------------------------------------------------------------------------
// Host-side orchestration
// ---------------------------------------------------------------------------
extern "C" void kernel_launch(void* const* d_in, const int* in_sizes, int n_in,
                              void* d_out, int out_size, void* d_ws, size_t ws_size,
                              hipStream_t stream) {
    const float* x     = (const float*)d_in[0];
    const int*   idx   = (const int*)d_in[1];
    const float* W_K   = (const float*)d_in[2];
    const float* W_V   = (const float*)d_in[3];
    const float* W_in  = (const float*)d_in[4];
    const float* W_out = (const float*)d_in[5];
    float* out = (float*)d_out;

    // ---- workspace layout (256B aligned slabs) ----
    char* w = (char*)d_ws;
    size_t off = 0;
    auto alloc = [&](size_t bytes) { char* p = w + off; off = (off + bytes + 255) & ~(size_t)255; return p; };
    unsigned int*   slots   = (unsigned int*)  alloc(4 * sizeof(unsigned int)); // maxabs K,V,in,out
    unsigned short* xbf     = (unsigned short*)alloc((size_t)MTOK * DM  * 2);   // x bf16
    unsigned short* WKb     = (unsigned short*)alloc((size_t)DM   * NKV * 2);   // W_K_sim bf16 [DM,NKV]
    unsigned short* WVTb    = (unsigned short*)alloc((size_t)NKV  * DM  * 2);   // W_V_simT bf16 [NKV,DM]
    unsigned short* WInb    = (unsigned short*)alloc((size_t)DM   * DFF * 2);   // W_in_sim bf16
    unsigned short* WOutb   = (unsigned short*)alloc((size_t)DFF  * DM  * 2);   // W_out_sim bf16
    float*          Sfull   = (float*)         alloc((size_t)MTOK * NKV * 4);   // scores f32
    unsigned short* Pbf     = (unsigned short*)alloc((size_t)MTOK * NKV * 2);   // probs bf16
    unsigned short* combbf  = (unsigned short*)alloc((size_t)MTOK * DM  * 2);   // comb bf16
    unsigned short* hidbf   = (unsigned short*)alloc((size_t)MTOK * DFF * 2);   // hidden bf16
    (void)ws_size; (void)n_in; (void)in_sizes; (void)out_size;

    // ---- 1) per-tensor max|w| ----
    init_slots<<<1, 32, 0, stream>>>(slots);
    maxabs_kernel<<<512, 256, 0, stream>>>(W_K,   (size_t)DM * NKV, slots + 0);
    maxabs_kernel<<<512, 256, 0, stream>>>(W_V,   (size_t)DM * NKV, slots + 1);
    maxabs_kernel<<<2048, 256, 0, stream>>>(W_in, (size_t)DM * DFF, slots + 2);
    maxabs_kernel<<<2048, 256, 0, stream>>>(W_out,(size_t)DFF * DM, slots + 3);

    // ---- 2) quantize weights -> bf16 (W_V transposed), convert x -> bf16 ----
    quant_bf16          <<<1024, 256, 0, stream>>>(W_K,   WKb,  (size_t)DM * NKV, slots + 0);
    quant_bf16_transpose<<<1024, 256, 0, stream>>>(W_V,   WVTb,                    slots + 1);
    quant_bf16          <<<4096, 256, 0, stream>>>(W_in,  WInb, (size_t)DM * DFF, slots + 2);
    quant_bf16          <<<4096, 256, 0, stream>>>(W_out, WOutb,(size_t)DFF * DM, slots + 3);
    cvt_bf16            <<<4096, 256, 0, stream>>>(x, xbf, (size_t)MTOK * DM);

    // ---- 3) full score matrix: S = x @ W_K_sim   [4096,2048]x[2048,256] ----
    gemm_bf16_wmma<false><<<dim3(NKV / 128, MTOK / 128), 256, 0, stream>>>(
        xbf, WKb, Sfull, MTOK, NKV, DM);

    // ---- 4) softmax over 16 gathered scores, scatter-add into dense P ----
    attn_softmax_scatter<<<MTOK / 256, 256, 0, stream>>>(Sfull, idx, Pbf);

    // ---- 5) comb = P @ W_V_simT   [4096,256]x[256,2048] ----
    gemm_bf16_wmma<true><<<dim3(DM / 128, MTOK / 128), 256, 0, stream>>>(
        Pbf, WVTb, combbf, MTOK, DM, NKV);

    // ---- 6) hidden = comb @ W_in_sim   [4096,2048]x[2048,8192] ----
    gemm_bf16_wmma<true><<<dim3(DFF / 128, MTOK / 128), 256, 0, stream>>>(
        combbf, WInb, hidbf, MTOK, DFF, DM);

    // ---- 7) out = hidden @ W_out_sim   [4096,8192]x[8192,2048] ----
    gemm_bf16_wmma<false><<<dim3(DM / 128, MTOK / 128), 256, 0, stream>>>(
        hidbf, WOutb, out, MTOK, DM, DFF);
}
